// SensorMultiHeadAttention_74844100100621
// MI455X (gfx1250) — matrix operand order repaired and checked by reading, not compiled, not run on hardware
//
#include <hip/hip_runtime.h>

typedef __attribute__((ext_vector_type(16))) _Float16 v16h;
typedef __attribute__((ext_vector_type(8)))  _Float16 v8h;
typedef __attribute__((ext_vector_type(4)))  _Float16 v4h;
typedef __attribute__((ext_vector_type(8)))  float    v8f;
typedef __attribute__((ext_vector_type(4)))  float    v4f;

#define D_MODEL 512
#define NSEQ    14
#define NPAD    16
#define DK      64
#define NHEADS  8
#define N2      (D_MODEL * D_MODEL)

// ---------------------------------------------------------------------------
// Kernel 0: convert Wq,Wk,Wv,Wo (f32 row-major [out,in]) -> f16 in workspace.
// ---------------------------------------------------------------------------
__global__ void cvt_weights_f16(const float* __restrict__ Wq,
                                const float* __restrict__ Wk,
                                const float* __restrict__ Wv,
                                const float* __restrict__ Wo,
                                _Float16* __restrict__ dst) {
  int i = blockIdx.x * blockDim.x + threadIdx.x;
  const int stride = gridDim.x * blockDim.x;
  for (; i < N2; i += stride) {
    dst[0 * N2 + i] = (_Float16)Wq[i];
    dst[1 * N2 + i] = (_Float16)Wk[i];
    dst[2 * N2 + i] = (_Float16)Wv[i];
    dst[3 * N2 + i] = (_Float16)Wo[i];
  }
}

// ---------------------------------------------------------------------------
// Fragment helpers (CDNA5 wave32 WMMA 16x16x32 f16 layouts, ISA 7.12.2)
// ---------------------------------------------------------------------------
__device__ __forceinline__ v16h make_v16(v8h lo, v8h hi) {
  v16h r;
#pragma unroll
  for (int i = 0; i < 8; ++i) { r[i] = lo[i]; r[8 + i] = hi[i]; }
  return r;
}

// A matrix 16x32 f16 from an LDS tile [NPAD][D_MODEL] at k window k0.
// lanes 0-15: row=lane, K [0..15]
// lanes 16-31: row=lane-16, K [16..31]
__device__ __forceinline__ v16h load_a_frag(const _Float16* s, int k0, int lane) {
  const int row = lane & 15;
  const int hs  = (lane >> 4) << 4;               // 0 or 16 elements
  v8h lo = *(const v8h*)(s + row * D_MODEL + k0 + hs);
  v8h hi = *(const v8h*)(s + row * D_MODEL + k0 + 8 + hs);
  return make_v16(lo, hi);
}

__device__ __forceinline__ v8f wmma_f16(v16h a, v16h b, v8f c) {
  return __builtin_amdgcn_wmma_f32_16x16x32_f16(false, a, false, b, (short)0, c,
                                                false, false);
}

// ---------------------------------------------------------------------------
// Fused attention kernel: one workgroup (8 wave32) handles G batch elements,
// so each weight fragment loaded into VGPRs feeds G WMMAs (L2 traffic / G).
// ---------------------------------------------------------------------------
template <int G>
__global__ __launch_bounds__(256)
void fused_sensor_mha(const float* __restrict__ x,
                      const float* __restrict__ bq, const float* __restrict__ bk,
                      const float* __restrict__ bv, const float* __restrict__ bo,
                      const _Float16* __restrict__ Wh,   // [4][512][512] f16
                      const float* __restrict__ scale_p,
                      const float* __restrict__ power_p,
                      const float* __restrict__ dist,    // [14][14] f32
                      float* __restrict__ out, int b_start) {
  __shared__ __align__(32) _Float16 sX[G][NPAD * D_MODEL];  // x tiles; reused for attn out
  __shared__ __align__(32) _Float16 sQ[G][NPAD * D_MODEL];
  __shared__ __align__(32) _Float16 sK[G][NPAD * D_MODEL];
  __shared__ __align__(32) _Float16 sV[G][NPAD * D_MODEL];

  const int tid  = threadIdx.x;
  const int lane = tid & 31;
  const int wave = tid >> 5;
  const int b0   = b_start + blockIdx.x * G;

  // ---- phase 1: stage x[b0..b0+G) -> f16 LDS, zero-pad rows 14,15 ----
#pragma unroll
  for (int g = 0; g < G; ++g) {
    const v4f* xb4 = (const v4f*)(x + (size_t)(b0 + g) * NSEQ * D_MODEL);
    for (int i = tid; i < NSEQ * D_MODEL / 4; i += 256) {
      v4f v = xb4[i];
      v4h h;
#pragma unroll
      for (int j = 0; j < 4; ++j) h[j] = (_Float16)v[j];
      *(v4h*)&sX[g][i * 4] = h;
    }
    v4h z = {};
    for (int i = tid; i < (NPAD - NSEQ) * D_MODEL / 4; i += 256)
      *(v4h*)&sX[g][NSEQ * D_MODEL + i * 4] = z;
  }
  __syncthreads();

  const int lncol = lane & 15;          // output column within 16-tile
  const int khalf = (lane >> 4) << 4;   // B-frag K offset: 0 or 16 elements
  const int rbase = (lane >> 4) << 3;   // C/D-frag row base: 0 or 8

  // ---- phase 2: Q,K,V = X @ W^T + b. 96 tiles, 12/wave, chunks of 3 ----
#pragma unroll
  for (int c = 0; c < 4; ++c) {
    const _Float16* bp[3];
    int mtxs[3], cols[3];
#pragma unroll
    for (int tt = 0; tt < 3; ++tt) {
      const int gt  = wave * 12 + c * 3 + tt;
      const int mtx = gt >> 5;                    // 0:Q 1:K 2:V
      const int n0  = (gt & 31) << 4;
      mtxs[tt] = mtx;
      cols[tt] = n0 + lncol;
      bp[tt]   = Wh + (size_t)mtx * N2 + (size_t)(n0 + lncol) * D_MODEL + khalf;
    }
    v8f acc[3][G];
    v8f zero = {};
#pragma unroll
    for (int tt = 0; tt < 3; ++tt)
#pragma unroll
      for (int g = 0; g < G; ++g) acc[tt][g] = zero;

    for (int k0 = 0; k0 < D_MODEL; k0 += 32) {
      v16h bf0 = *(const v16h*)(bp[0] + k0);
      v16h bf1 = *(const v16h*)(bp[1] + k0);
      v16h bf2 = *(const v16h*)(bp[2] + k0);
#pragma unroll
      for (int g = 0; g < G; ++g) {
        v16h a = load_a_frag(sX[g], k0, lane);
        acc[0][g] = wmma_f16(a, bf0, acc[0][g]);
        acc[1][g] = wmma_f16(a, bf1, acc[1][g]);
        acc[2][g] = wmma_f16(a, bf2, acc[2][g]);
      }
    }
#pragma unroll
    for (int tt = 0; tt < 3; ++tt) {
      _Float16* base     = (mtxs[tt] == 0) ? &sQ[0][0]
                         : (mtxs[tt] == 1) ? &sK[0][0] : &sV[0][0];
      const float* biasp = (mtxs[tt] == 0) ? bq : (mtxs[tt] == 1) ? bk : bv;
      const float bb = biasp[cols[tt]];
#pragma unroll
      for (int g = 0; g < G; ++g)
#pragma unroll
        for (int i = 0; i < 8; ++i)
          base[g * NPAD * D_MODEL + (rbase + i) * D_MODEL + cols[tt]] =
              (_Float16)(acc[tt][g][i] + bb);
    }
  }
  __syncthreads();

  // ---- phase 3: attention (f32 VALU). wave == head, lane == query row ----
  {
    v4h z = {};
#pragma unroll
    for (int g = 0; g < G; ++g)
      for (int i = tid; i < (NPAD - NSEQ) * D_MODEL / 4; i += 256)
        *(v4h*)&sX[g][NSEQ * D_MODEL + i * 4] = z;
  }
  {
    const int h = wave;
    const int i = lane;
    if (i < NSEQ) {
      const float scl = scale_p[0];
      const float pwr = power_p[0];
      float pr[NSEQ];                       // distance prior, batch-invariant
#pragma unroll
      for (int j = 0; j < NSEQ; ++j) {
        const float dd = dist[i * NSEQ + j];
        const float pw = (dd > 0.f) ? expf(pwr * logf(dd)) : 0.f;
        pr[j] = scl / (1.f + pw) * 0.125f;  // fold in 1/sqrt(64)
      }
#pragma unroll
      for (int g = 0; g < G; ++g) {
        float q[DK];
#pragma unroll
        for (int d = 0; d < DK; ++d) q[d] = (float)sQ[g][i * D_MODEL + h * DK + d];
        float sc[NSEQ];
        float mx = -3.0e38f;
        for (int j = 0; j < NSEQ; ++j) {
          float s = 0.f;
#pragma unroll
          for (int d = 0; d < DK; ++d)
            s += q[d] * (float)sK[g][j * D_MODEL + h * DK + d];
          s *= pr[j];
          sc[j] = s;
          mx = fmaxf(mx, s);
        }
        float sum = 0.f;
        for (int j = 0; j < NSEQ; ++j) { sc[j] = expf(sc[j] - mx); sum += sc[j]; }
        const float inv = 1.f / sum;
        for (int d = 0; d < DK; ++d) {
          float o = 0.f;
          for (int j = 0; j < NSEQ; ++j)
            o += sc[j] * (float)sV[g][j * D_MODEL + h * DK + d];
          sX[g][i * D_MODEL + h * DK + d] = (_Float16)(o * inv);
        }
      }
    }
  }
  __syncthreads();

  // ---- phase 4: out = attn_out @ Wo^T + bo. 32 n-tiles, 4/wave, chunks of 2 ----
  const _Float16* Woh = Wh + (size_t)3 * N2;
#pragma unroll
  for (int c = 0; c < 2; ++c) {
    const _Float16* bp[2];
    int cols[2];
#pragma unroll
    for (int tt = 0; tt < 2; ++tt) {
      const int n0 = (wave * 4 + c * 2 + tt) << 4;
      cols[tt] = n0 + lncol;
      bp[tt]   = Woh + (size_t)(n0 + lncol) * D_MODEL + khalf;
    }
    v8f acc[2][G];
    v8f zero = {};
#pragma unroll
    for (int tt = 0; tt < 2; ++tt)
#pragma unroll
      for (int g = 0; g < G; ++g) acc[tt][g] = zero;

    for (int k0 = 0; k0 < D_MODEL; k0 += 32) {
      v16h bf0 = *(const v16h*)(bp[0] + k0);
      v16h bf1 = *(const v16h*)(bp[1] + k0);
#pragma unroll
      for (int g = 0; g < G; ++g) {
        v16h a = load_a_frag(sX[g], k0, lane);
        acc[0][g] = wmma_f16(a, bf0, acc[0][g]);
        acc[1][g] = wmma_f16(a, bf1, acc[1][g]);
      }
    }
#pragma unroll
    for (int tt = 0; tt < 2; ++tt) {
      const float bb = bo[cols[tt]];
#pragma unroll
      for (int g = 0; g < G; ++g) {
#pragma unroll
        for (int i = 0; i < 8; ++i) {
          const int m = rbase + i;
          if (m < NSEQ)
            out[((size_t)(b0 + g) * NSEQ + m) * D_MODEL + cols[tt]] =
                acc[tt][g][i] + bb;
        }
      }
    }
  }
}

// ---------------------------------------------------------------------------
extern "C" void kernel_launch(void* const* d_in, const int* in_sizes, int n_in,
                              void* d_out, int out_size, void* d_ws, size_t ws_size,
                              hipStream_t stream) {
  const float* x     = (const float*)d_in[0];
  const float* Wq    = (const float*)d_in[1];
  const float* bq    = (const float*)d_in[2];
  const float* Wk    = (const float*)d_in[3];
  const float* bk    = (const float*)d_in[4];
  const float* Wv    = (const float*)d_in[5];
  const float* bv    = (const float*)d_in[6];
  const float* Wo    = (const float*)d_in[7];
  const float* bo    = (const float*)d_in[8];
  const float* scale = (const float*)d_in[9];
  const float* power = (const float*)d_in[10];
  const float* dist  = (const float*)d_in[11];

  _Float16* Wh = (_Float16*)d_ws;                 // 4 * 512*512 * 2B = 2 MB scratch
  const int Bn = in_sizes[0] / (NSEQ * D_MODEL);  // 8192

  cvt_weights_f16<<<512, 256, 0, stream>>>(Wq, Wk, Wv, Wo, Wh);

  constexpr int G = 4;
  const int nMain = Bn / G;
  const int rem   = Bn - nMain * G;
  if (nMain > 0)
    fused_sensor_mha<G><<<nMain, 256, 0, stream>>>(x, bq, bk, bv, bo, Wh,
                                                   scale, power, dist,
                                                   (float*)d_out, 0);
  if (rem > 0)
    fused_sensor_mha<1><<<rem, 256, 0, stream>>>(x, bq, bk, bv, bo, Wh,
                                                 scale, power, dist,
                                                 (float*)d_out, nMain * G);
}
